// AgentCuriosity_77103252898059
// MI455X (gfx1250) — matrix operand order repaired and checked by reading, not compiled, not run on hardware
//
#include <hip/hip_runtime.h>

// ---------------------------------------------------------------------------
// Fused BRIMs scan for MI455X (gfx1250, wave32, WMMA).
//  - One persistent kernel: 16 WGs x 16 batch rows run the full T=128 scan
//    (batch rows independent -> no grid sync, no per-step launches).
//  - Weights pre-swizzled ONCE per launch into d_ws as f16 tiles in the
//    v_wmma_f32_16x16x32_f16 B-fragment order: one 32B vector load per lane.
//  - GEMM-A activations kept f16 row-major in LDS: 2x ds_load_b128/fragment.
//  - Long K-loops software-pipeline the B loads (wrap-around prefetch) so
//    each WMMA consumes a fragment that was already in flight.
//  - Output heads (512->26/26/1) are WMMA on a concatenated padded 512x64
//    head matrix instead of ~434K scalar FMAs per WG per step.
//  - Weights (~2.6 MB f16) stay L2-resident (192 MB L2); HBM traffic is the
//    state tensor + outputs (~32 MB) -> ~1.4 us at 23.3 TB/s.  The kernel is
//    latency/L2-bound, so the goal is minimal instructions per WMMA and
//    loads that are issued ahead of their consumers.
// ---------------------------------------------------------------------------

#define T_STEPS 128
#define BATCH   256
#define ROWS    16
#define NGRP    (BATCH / ROWS)
#define NTHREADS 512
#define NWAVES  (NTHREADS / 32)
#define NHID_   512
#define NINP_   512
#define SDIM    172
#define NB_     8
#define DD      64
#define GATES   256
#define HH      4
#define DH_     16
#define AA      26

typedef __attribute__((ext_vector_type(16))) _Float16 v16h;
typedef __attribute__((ext_vector_type(8)))  _Float16 v8h;
typedef __attribute__((ext_vector_type(8)))  float    v8f;

#define WMMA_F16(a, b, c) \
  __builtin_amdgcn_wmma_f32_16x16x32_f16(false, (a), false, (b), (short)0, (c), false, false)

// ---- swizzled-weight arena offsets inside d_ws (in _Float16 elements) ----
// tile = 32(K) x 16(N) = 512 halves, stored as [lane][16] (one 32B run/lane)
#define SZ_ENC   (6 * 32 * 512)            // 172->192 K-pad, 32 N-tiles
#define SZ_WQ_L  (16 * 32 * 512)
#define SZ_WK_L  (16 * 4 * 512)
#define SZ_WI_M  (2 * 16 * 512)            // one (layer, block) 64x256 matrix
#define SZ_CQ_L  (2 * 4 * 512)
#define SZ_HD    (16 * 4 * 512)            // concatenated heads 512 x 64(pad)
#define OFF_ENC  0
#define OFF_WQ   (OFF_ENC + SZ_ENC)
#define OFF_WK   (OFF_WQ + 2 * SZ_WQ_L)
#define OFF_WV   (OFF_WK + 2 * SZ_WK_L)
#define OFF_WI   (OFF_WV + 2 * SZ_WK_L)
#define OFF_WH   (OFF_WI + 16 * SZ_WI_M)
#define OFF_CQ   (OFF_WH + 16 * SZ_WI_M)
#define OFF_CK   (OFF_CQ + 2 * SZ_CQ_L)
#define OFF_CV   (OFF_CK + 2 * SZ_CQ_L)
#define OFF_CO   (OFF_CV + 2 * SZ_CQ_L)
#define OFF_HD   (OFF_CO + 2 * SZ_CQ_L)
#define WS_HALFS (OFF_HD + SZ_HD)          // 1,343,488 halves = 2.56 MB

__device__ __forceinline__ float sigmoidf_(float x) { return 1.0f / (1.0f + __expf(-x)); }

// A fragment from f32 global (encoder only), K-guarded.
__device__ __forceinline__ v16h load_a_f32(const float* A, int ld, int k0, int kmax, int lane) {
  int r = lane & 15, hi = lane >> 4;
  v16h a;
#pragma unroll
  for (int j = 0; j < 8; ++j) {
    int kA = k0 + hi * 8 + j;
    int kB = k0 + 16 + hi * 8 + j;
    a[j]     = (_Float16)(kA < kmax ? A[(size_t)r * ld + kA] : 0.0f);
    a[j + 8] = (_Float16)(kB < kmax ? A[(size_t)r * ld + kB] : 0.0f);
  }
  return a;
}

// A fragment from f16 row-major LDS: two contiguous 8-half runs per lane.
__device__ __forceinline__ v16h load_a_h(const _Float16* A, int ld, int k0, int lane) {
  int r = lane & 15, hi = lane >> 4;
  const _Float16* p = A + (size_t)r * ld + k0 + hi * 8;
  v8h lo = *(const v8h*)p;
  v8h hh = *(const v8h*)(p + 16);
  return __builtin_shufflevector(lo, hh, 0, 1, 2, 3, 4, 5, 6, 7,
                                 8, 9, 10, 11, 12, 13, 14, 15);
}

// B fragment from the pre-swizzled f16 arena: one 32B vector load per lane.
__device__ __forceinline__ v16h load_b_sw(const _Float16* Wsw, int tilesN, int tk, int tn, int lane) {
  return *(const v16h*)(Wsw + (((size_t)(tk * tilesN + tn)) << 9) + (lane << 4));
}

// D store: 16x16 f32 tile; row m = hi*8+j, col = lane%16.
__device__ __forceinline__ void store_d(float* Dst, int ld, int n0, v8f acc, int lane) {
  int r = lane & 15, hi = lane >> 4;
#pragma unroll
  for (int j = 0; j < 8; ++j) Dst[(size_t)(hi * 8 + j) * ld + n0 + r] = acc[j];
}

// ---------------------------------------------------------------------------
// One-shot weight swizzles (row-major f32 -> f16 WMMA B-fragment tiles).
// ---------------------------------------------------------------------------
extern "C" __global__ void swizzle_w_f16(const float* __restrict__ src,
                                         _Float16* __restrict__ dst,
                                         int K, int N, int nmat) {
  int KP = (K + 31) & ~31;
  int tilesK = KP >> 5, tilesN = N >> 4;
  size_t tilesPerMat = (size_t)tilesK * tilesN;
  size_t total = (size_t)nmat * tilesPerMat * 32;
  for (size_t idx = (size_t)blockIdx.x * blockDim.x + threadIdx.x; idx < total;
       idx += (size_t)gridDim.x * blockDim.x) {
    int lane = (int)(idx & 31);
    size_t tile = idx >> 5;
    int m = (int)(tile / tilesPerMat);
    size_t tin = tile - (size_t)m * tilesPerMat;
    int tk = (int)(tin / tilesN), tn = (int)(tin % tilesN);
    int n = lane & 15, hi = lane >> 4;
    const float* S = src + (size_t)m * K * N;
    _Float16* Dp = dst + (tile << 9) + (lane << 4);
#pragma unroll
    for (int e = 0; e < 16; ++e) {
      int k = tk * 32 + hi * 8 + e + (e >= 8 ? 8 : 0);
      Dp[e] = (_Float16)(k < K ? S[(size_t)k * N + tn * 16 + n] : 0.0f);
    }
  }
}

// Heads: [Wmu(512x26) | Wstd(512x26) | Wc(512x1) | pad] -> 512x64 f16 tiles.
extern "C" __global__ void swizzle_head_f16(const float* __restrict__ Wmu,
                                            const float* __restrict__ Wstd,
                                            const float* __restrict__ Wc,
                                            _Float16* __restrict__ dst) {
  for (int idx = blockIdx.x * blockDim.x + threadIdx.x; idx < 16 * 4 * 32;
       idx += gridDim.x * blockDim.x) {
    int lane = idx & 31;
    int tile = idx >> 5;
    int tk = tile >> 2, tn = tile & 3;
    int n = lane & 15, hi = lane >> 4;
    _Float16* Dp = dst + ((size_t)tile << 9) + (lane << 4);
#pragma unroll
    for (int e = 0; e < 16; ++e) {
      int k = tk * 32 + hi * 8 + e + (e >= 8 ? 8 : 0);
      int col = tn * 16 + n;
      float v = 0.0f;
      if (col < AA) v = Wmu[(size_t)k * AA + col];
      else if (col < 2 * AA) v = Wstd[(size_t)k * AA + (col - AA)];
      else if (col == 2 * AA) v = Wc[k];
      Dp[e] = (_Float16)v;
    }
  }
}

extern "C" __global__ __launch_bounds__(NTHREADS, 1)
void brims_scan_wmma(const _Float16* __restrict__ wsw, const float* __restrict__ state,
                     const float* __restrict__ hx0, const float* __restrict__ cx0,
                     const float* __restrict__ encB, const float* __restrict__ bl,
                     const float* __restrict__ bmu, const float* __restrict__ bstd,
                     const float* __restrict__ bc, float* __restrict__ out) {
  const int tid  = threadIdx.x;
  const int lane = tid & 31;
  const int wave = tid >> 5;
  const int row0 = blockIdx.x * ROWS;

  // ---- LDS: 205.8 KB f32 + 80 KB f16 = 287.7 KB of the WGP's 320 KB ----
  extern __shared__ float smem[];
  float* Q   = smem;                 // q -> later qc           (16x512 f32)
  float* KC  = Q + 8192;             // kc -> later comm
  float* VC  = KC + 8192;            // vc
  float* CN  = VC + 8192;            // c_new
  float* HNf = CN + 8192;            // h_new (f32 for masking)
  float* G   = HNf + 8192;           // gates ping-pong (2 x 16x256)
  float* KX  = G + 2 * 4096;         // input key   (16x64)
  float* VX  = KX + 1024;            // input value (16x64)
  float* SC  = VX + 1024;            // slot-attention score (16x8)
  float* MK  = SC + 128;             // top-k mask (16x8)
  _Float16* XH   = (_Float16*)(MK + 128);  // x / hidden        (16x512 f16)
  _Float16* BH16 = XH + 8192;        // staged h (GEMM A)
  _Float16* INP  = BH16 + 8192;      // slot-attention output
  _Float16* HN16 = INP + 8192;       // h_new (GEMM A)
  _Float16* AO   = HN16 + 8192;      // comm-attention output (GEMM A)

  // ---- d_out regions: mu | log_std | value | hx | cx (hx/cx double as state)
  float* outMu  = out;
  float* outLs  = outMu + (size_t)T_STEPS * BATCH * AA;
  float* outVal = outLs + (size_t)T_STEPS * BATCH * AA;
  float* hxS    = outVal + (size_t)T_STEPS * BATCH;
  float* cxS    = hxS + (size_t)2 * BATCH * NHID_;

  for (int i = tid; i < 2 * ROWS * NHID_; i += NTHREADS) {
    int l  = i / (ROWS * NHID_);
    int rr = (i / NHID_) % ROWS;
    int j  = i % NHID_;
    size_t g = (size_t)l * BATCH * NHID_ + (size_t)(row0 + rr) * NHID_ + j;
    hxS[g] = hx0[g];
    cxS[g] = cx0[g];
  }
  __syncthreads();

  for (int t = 0; t < T_STEPS; ++t) {
    // ---- Encoder: XH = relu(state[t] @ encW + encB), K guarded at 172 ----
    {
      const float* Ast = state + ((size_t)t * BATCH + row0) * SDIM;
      const _Float16* ENCs = wsw + OFF_ENC;
      for (int n0 = wave * 16; n0 < NINP_; n0 += NWAVES * 16) {
        int tn = n0 >> 4;
        v8f acc = {};
        v16h bb = load_b_sw(ENCs, 32, 0, tn, lane);
        for (int tk = 0; tk < 6; ++tk) {
          v16h a  = load_a_f32(Ast, SDIM, tk * 32, SDIM, lane);
          v16h bn = load_b_sw(ENCs, 32, tk < 5 ? tk + 1 : 0, tn, lane);  // prefetch
          acc = WMMA_F16(a, bb, acc);
          bb = bn;
        }
        int r = lane & 15, hi = lane >> 4;
#pragma unroll
        for (int j = 0; j < 8; ++j)
          XH[(size_t)(hi * 8 + j) * NHID_ + n0 + r] =
              (_Float16)fmaxf(acc[j] + encB[n0 + r], 0.0f);
      }
    }
    __syncthreads();

    for (int l = 0; l < 2; ++l) {
      // Stage h (f16) for the q / Wh GEMMs.
      for (int i = tid; i < ROWS * NHID_; i += NTHREADS) {
        int b = i >> 9, j = i & (NHID_ - 1);
        BH16[i] = (_Float16)hxS[(size_t)l * BATCH * NHID_ + (size_t)(row0 + b) * NHID_ + j];
      }
      __syncthreads();

      const _Float16* WQs = wsw + OFF_WQ + (size_t)l * SZ_WQ_L;
      const _Float16* WKs = wsw + OFF_WK + (size_t)l * SZ_WK_L;
      const _Float16* WVs = wsw + OFF_WV + (size_t)l * SZ_WK_L;

      // ---- Phase 1+2 fused: KX/VX = x@Wk/Wv (8 tiles), Q = h@Wq (32 tiles),
      //      with software-pipelined B prefetch over the 16 K-chunks.
      for (int job = wave; job < 40; job += NWAVES) {
        if (job < 8) {
          int which = job >> 2;
          int tn = job & 3;
          const _Float16* Bw = which ? WVs : WKs;
          float* Dst = which ? VX : KX;
          v8f acc = {};
          v16h bb = load_b_sw(Bw, 4, 0, tn, lane);
          for (int tk = 0; tk < 16; ++tk) {
            v16h a  = load_a_h(XH, NHID_, tk * 32, lane);
            v16h bn = load_b_sw(Bw, 4, (tk + 1) & 15, tn, lane);  // prefetch
            acc = WMMA_F16(a, bb, acc);
            bb = bn;
          }
          store_d(Dst, DD, tn * 16, acc, lane);
        } else {
          int tn = job - 8;
          v8f acc = {};
          v16h bb = load_b_sw(WQs, 32, 0, tn, lane);
          for (int tk = 0; tk < 16; ++tk) {
            v16h a  = load_a_h(BH16, NHID_, tk * 32, lane);
            v16h bn = load_b_sw(WQs, 32, (tk + 1) & 15, tn, lane);  // prefetch
            acc = WMMA_F16(a, bb, acc);
            bb = bn;
          }
          store_d(Q, NHID_, tn * 16, acc, lane);
        }
      }
      __syncthreads();

      // ---- Phase 3: slot attention (null slot -> sigmoid), inp, top-k mask.
      for (int i = tid; i < ROWS * NB_; i += NTHREADS) {
        int b = i >> 3, n = i & 7;
        float s = 0.0f;
        const float* qv = Q + (size_t)b * NHID_ + n * DD;
        const float* kv = KX + (size_t)b * DD;
        for (int d = 0; d < DD; ++d) s += qv[d] * kv[d];
        SC[i] = sigmoidf_(s * 0.125f);                     // 1/sqrt(DK)
      }
      __syncthreads();
      if (tid < ROWS) {
        float sc[NB_], mk[NB_];
        for (int n = 0; n < NB_; ++n) { sc[n] = SC[tid * NB_ + n]; mk[n] = 0.0f; }
        for (int k = 0; k < 4; ++k) {
          int bi = 0; float bv = -3.4e38f;
          for (int n = 0; n < NB_; ++n)
            if (mk[n] == 0.0f && sc[n] > bv) { bv = sc[n]; bi = n; }
          mk[bi] = 1.0f;
        }
        for (int n = 0; n < NB_; ++n) MK[tid * NB_ + n] = mk[n];
      }
      for (int i = tid; i < ROWS * NHID_; i += NTHREADS) {
        int b = i >> 9, j = i & (NHID_ - 1);
        int n = j >> 6, d = j & 63;
        INP[i] = (_Float16)(SC[b * NB_ + n] * VX[(size_t)b * DD + d]);
      }
      __syncthreads();

      // ---- Phase 4: per-block gates (WMMA) + LSTM (VALU), ping-pong G so
      //      block n+1's WMMA overlaps block n's LSTM (one barrier/block).
      const float* bll = bl + (size_t)l * NB_ * GATES;
      for (int n = 0; n < NB_; ++n) {
        float* Gn = G + (n & 1) * (ROWS * GATES);
        const _Float16* WIb = wsw + OFF_WI + (size_t)(l * NB_ + n) * SZ_WI_M;
        const _Float16* WHb = wsw + OFF_WH + (size_t)(l * NB_ + n) * SZ_WI_M;
        for (int n0 = wave * 16; n0 < GATES; n0 += NWAVES * 16) {
          v8f acc = {};
#pragma unroll
          for (int k0 = 0; k0 < DD; k0 += 32) {
            v16h a = load_a_h(INP + n * DD, NHID_, k0, lane);
            v16h b = load_b_sw(WIb, 16, k0 >> 5, n0 >> 4, lane);
            acc = WMMA_F16(a, b, acc);
            a = load_a_h(BH16 + n * DD, NHID_, k0, lane);
            b = load_b_sw(WHb, 16, k0 >> 5, n0 >> 4, lane);
            acc = WMMA_F16(a, b, acc);
          }
          int r = lane & 15, hi = lane >> 4;
#pragma unroll
          for (int j = 0; j < 8; ++j)
            Gn[(size_t)(hi * 8 + j) * GATES + n0 + r] = acc[j] + bll[n * GATES + n0 + r];
        }
        __syncthreads();
        for (int i = tid; i < ROWS * DD; i += NTHREADS) {
          int b = i >> 6, d = i & 63;
          float ig = Gn[(size_t)b * GATES + d];
          float fg = Gn[(size_t)b * GATES + DD + d];
          float gg = Gn[(size_t)b * GATES + 2 * DD + d];
          float og = Gn[(size_t)b * GATES + 3 * DD + d];
          float cold = cxS[(size_t)l * BATCH * NHID_ + (size_t)(row0 + b) * NHID_ + n * DD + d];
          float cn = sigmoidf_(fg) * cold + sigmoidf_(ig) * tanhf(gg);
          float hn = sigmoidf_(og) * tanhf(cn);
          CN[(size_t)b * NHID_ + n * DD + d] = cn;
          HNf[(size_t)b * NHID_ + n * DD + d] = hn;
          HN16[(size_t)b * NHID_ + n * DD + d] = (_Float16)hn;
        }
      }
      __syncthreads();

      // ---- Phase 5: qc/kc/vc = h_new(block) @ Cq/Ck/Cv -> Q/KC/VC (f32).
      const _Float16* CQs = wsw + OFF_CQ + (size_t)l * SZ_CQ_L;
      const _Float16* CKs = wsw + OFF_CK + (size_t)l * SZ_CQ_L;
      const _Float16* CVs = wsw + OFF_CV + (size_t)l * SZ_CQ_L;
      for (int job = wave; job < 96; job += NWAVES) {
        int mat = job / 32;
        int rem = job - mat * 32;
        int n = rem >> 2, n0 = (rem & 3) * 16;
        const _Float16* Bw = (mat == 0) ? CQs : (mat == 1) ? CKs : CVs;
        float* Dst = (mat == 0) ? Q : (mat == 1) ? KC : VC;
        v8f acc = {};
#pragma unroll
        for (int k0 = 0; k0 < DD; k0 += 32) {
          v16h a = load_a_h(HN16 + n * DD, NHID_, k0, lane);
          v16h b = load_b_sw(Bw, 4, k0 >> 5, n0 >> 4, lane);
          acc = WMMA_F16(a, b, acc);
        }
        store_d(Dst + n * DD, NHID_, n0, acc, lane);
      }
      __syncthreads();

      // ---- Phase 6: 8x8 per-head comm attention (VALU) -> AO (f16).
      for (int i = tid; i < ROWS * HH * NB_; i += NTHREADS) {
        int b = i >> 5;
        int rem = i & 31;
        int hh = rem >> 3, n = rem & 7;
        const float* qq = Q + (size_t)b * NHID_ + n * DD + hh * DH_;
        float sc[NB_]; float mx = -3.4e38f;
        for (int m = 0; m < NB_; ++m) {
          const float* kk = KC + (size_t)b * NHID_ + m * DD + hh * DH_;
          float s = 0.0f;
#pragma unroll
          for (int d = 0; d < DH_; ++d) s += qq[d] * kk[d];
          sc[m] = s * 0.25f;                               // 1/sqrt(DH)
          mx = fmaxf(mx, sc[m]);
        }
        float den = 0.0f;
        for (int m = 0; m < NB_; ++m) { sc[m] = __expf(sc[m] - mx); den += sc[m]; }
        float inv = 1.0f / den;
        for (int d = 0; d < DH_; ++d) {
          float o = 0.0f;
          for (int m = 0; m < NB_; ++m)
            o += sc[m] * VC[(size_t)b * NHID_ + m * DD + hh * DH_ + d];
          AO[(size_t)b * NHID_ + n * DD + hh * DH_ + d] = (_Float16)(o * inv);
        }
      }
      __syncthreads();

      // ---- Phase 7: comm = attn_out(block) @ Co -> KC (kc is dead).
      const _Float16* COs = wsw + OFF_CO + (size_t)l * SZ_CQ_L;
      for (int job = wave; job < 32; job += NWAVES) {
        int n = job >> 2, n0 = (job & 3) * 16;
        v8f acc = {};
#pragma unroll
        for (int k0 = 0; k0 < DD; k0 += 32) {
          v16h a = load_a_h(AO + n * DD, NHID_, k0, lane);
          v16h b = load_b_sw(COs, 4, k0 >> 5, n0 >> 4, lane);
          acc = WMMA_F16(a, b, acc);
        }
        store_d(KC + n * DD, NHID_, n0, acc, lane);
      }
      __syncthreads();

      // ---- Phase 8: masked update; write state; x_next/hidden (f16).
      for (int i = tid; i < ROWS * NHID_; i += NTHREADS) {
        int b = i >> 9, j = i & (NHID_ - 1);
        int n = j >> 6;
        float m = MK[b * NB_ + n];
        size_t g = (size_t)l * BATCH * NHID_ + (size_t)(row0 + b) * NHID_ + j;
        float hout = (m > 0.5f) ? (HNf[i] + KC[i]) : hxS[g];
        float cout = (m > 0.5f) ? CN[i] : cxS[g];
        hxS[g] = hout;
        cxS[g] = cout;
        XH[i] = (_Float16)hout;   // next layer's x / this step's hidden
      }
      __syncthreads();
    }  // layers

    // ---- Output heads via WMMA: [mu|log_std|value|pad] = hidden @ HD.
    {
      const _Float16* HDs = wsw + OFF_HD;
      for (int job = wave; job < 4; job += NWAVES) {
        v8f acc = {};
        v16h bb = load_b_sw(HDs, 4, 0, job, lane);
        for (int tk = 0; tk < 16; ++tk) {
          v16h a  = load_a_h(XH, NHID_, tk * 32, lane);
          v16h bn = load_b_sw(HDs, 4, (tk + 1) & 15, job, lane);  // prefetch
          acc = WMMA_F16(a, bb, acc);
          bb = bn;
        }
        int r = lane & 15, hi = lane >> 4;
        int col = job * 16 + r;
#pragma unroll
        for (int j = 0; j < 8; ++j) {
          int b = hi * 8 + j;
          size_t orow = (size_t)t * BATCH + row0 + b;
          float s = acc[j];
          if (col < AA) outMu[orow * AA + col] = s + bmu[col];
          else if (col < 2 * AA)
            outLs[orow * AA + (col - AA)] = fminf(fmaxf(s + bstd[col - AA], -20.0f), 2.0f);
          else if (col == 2 * AA) outVal[orow] = s + bc[0];
        }
      }
    }
    __syncthreads();  // XH reused next timestep
  }  // t
}

// f32: Q,KC,VC,CN,HNf (5x8192) + G (2x4096) + KX,VX (2048) + SC,MK (256)
// f16: XH,BH16,INP,HN16,AO (5x8192)
#define SMEM_BYTES ((5 * 8192 + 2 * 4096 + 2048 + 256) * 4 + (5 * 8192) * 2)  // 287,744

extern "C" void kernel_launch(void* const* d_in, const int* in_sizes, int n_in,
                              void* d_out, int out_size, void* d_ws, size_t ws_size,
                              hipStream_t stream) {
  (void)in_sizes; (void)n_in; (void)ws_size; (void)out_size;
  const float* state = (const float*)d_in[0];
  const float* hx0   = (const float*)d_in[1];
  const float* cx0   = (const float*)d_in[2];
  const float* encW  = (const float*)d_in[3];
  const float* encB  = (const float*)d_in[4];
  const float* Wq    = (const float*)d_in[5];
  const float* Wk    = (const float*)d_in[6];
  const float* Wv    = (const float*)d_in[7];
  const float* Wi    = (const float*)d_in[8];
  const float* Wh    = (const float*)d_in[9];
  const float* bl    = (const float*)d_in[10];
  const float* Cq    = (const float*)d_in[11];
  const float* Ck    = (const float*)d_in[12];
  const float* Cv    = (const float*)d_in[13];
  const float* Co    = (const float*)d_in[14];
  const float* Wmu   = (const float*)d_in[15];
  const float* bmu   = (const float*)d_in[16];
  const float* Wstd  = (const float*)d_in[17];
  const float* bstd  = (const float*)d_in[18];
  const float* Wc    = (const float*)d_in[19];
  const float* bc    = (const float*)d_in[20];
  float* out = (float*)d_out;
  _Float16* wsw = (_Float16*)d_ws;   // needs WS_HALFS*2 = 2.56 MB of scratch

  // One-shot weight swizzle into WMMA B-fragment order (deterministic, cheap).
  auto SW = [&](const float* src, size_t off, int K, int N, int nmat) {
    swizzle_w_f16<<<dim3(256), dim3(256), 0, stream>>>(src, wsw + off, K, N, nmat);
  };
  SW(encW, OFF_ENC, SDIM, NINP_, 1);
  SW(Wq,   OFF_WQ,  NHID_, NHID_, 2);
  SW(Wk,   OFF_WK,  NINP_, DD, 2);
  SW(Wv,   OFF_WV,  NINP_, DD, 2);
  SW(Wi,   OFF_WI,  DD, GATES, 16);
  SW(Wh,   OFF_WH,  DD, GATES, 16);
  SW(Cq,   OFF_CQ,  DD, DD, 2);
  SW(Ck,   OFF_CK,  DD, DD, 2);
  SW(Cv,   OFF_CV,  DD, DD, 2);
  SW(Co,   OFF_CO,  DD, DD, 2);
  swizzle_head_f16<<<dim3(8), dim3(256), 0, stream>>>(Wmu, Wstd, Wc, wsw + OFF_HD);

  (void)hipFuncSetAttribute((const void*)brims_scan_wmma,
                            hipFuncAttributeMaxDynamicSharedMemorySize, (int)SMEM_BYTES);
  brims_scan_wmma<<<dim3(NGRP), dim3(NTHREADS), SMEM_BYTES, stream>>>(
      wsw, state, hx0, cx0, encB, bl, bmu, bstd, bc, out);
}
